// Network_24713241821973
// MI455X (gfx1250) — matrix operand order repaired and checked by reading, not compiled, not run on hardware
//
#include <hip/hip_runtime.h>
#include <stdint.h>
#include <stddef.h>

// ---------------------------------------------------------------------------
// Pairwise MLP  dro[b,i,j] = W3^T relu(W2 relu(hA[b,i]+hB[b,j]) + b2) + b3
// B=2 N=512 D=160 H=320.  Layer-2 GEMM (107 GFLOP) on v_wmma_f32_16x16x32_f16.
// v2: two i-rows per wave share every B fragment (1 ds_load_b128 per WMMA),
//     LDS layout [nt][kt][lane][v] so the kt loop uses DS immediate offsets.
// ---------------------------------------------------------------------------

typedef _Float16 f16;
typedef __attribute__((ext_vector_type(16))) _Float16 v16h;
typedef __attribute__((ext_vector_type(8)))  float    v8f;

#define B_   2
#define N_   512
#define E_   128
#define L_   32
#define D_   160      // E_+L_
#define H_   320      // 2*D_
#define KP_  (H_/2)   // 160 packed f16 pairs per hidden row
#define KT_  (H_/32)  // 10 k-tiles of 32
#define NT_  (H_/16)  // 20 n-tiles of 16
#define W2PACK_U32 (H_*H_/2)        // 51200 u32 = 200 KB
#define HAB  (B_*N_*KP_)            // 163840 u32 per hA/hB buffer

#define BLK  256                    // 8 waves / workgroup (512-VGPR budget)
#define TPW  4                      // i-pair tiles per wave
#define NTILES2 (B_*(N_/2)*(N_/16)) // 16384 (i-pair, jb) tiles
#define GRID (NTILES2/((BLK/32)*TPW)) // 512

union PF { _Float16 h[2]; uint32_t u; };

// --- kernel 0: swizzle W2 (f32 row-major [H][H]) into WMMA B-fragment layout.
// 16-bit B 32x16 tile: lane = ((k%32)&8 ? 16:0) + (n%16); VGPR v = ((k%32)/16)*4 + (k%8)/2,
// low half = even k.  Stored as [nt][kt][lane][v] u32 (kt innermost so the
// compute kernel's kt loop fits in 16-bit DS immediate offsets).
__global__ void pack_w2_kernel(const float* __restrict__ W2,
                               uint32_t* __restrict__ w2pk) {
  int t = blockIdx.x * 256 + threadIdx.x;
  if (t >= W2PACK_U32) return;
  int k = (t / H_) * 2;
  int n = t % H_;
  PF p;
  p.h[0] = (_Float16)W2[(size_t)k * H_ + n];
  p.h[1] = (_Float16)W2[(size_t)(k + 1) * H_ + n];
  int kt   = k >> 5;
  int kin  = k & 31;
  int lane = ((kin & 8) ? 16 : 0) + (n & 15);
  int v    = ((kin >> 4) << 2) + ((kin & 7) >> 1);
  int nt   = n >> 4;
  w2pk[((nt * KT_ + kt) << 8) + lane * 8 + v] = p.u;
}

// --- kernel 1: hA = PhiA @ W1[:D], hB = PhiB @ W1[D:] + b1, packed f16 pairs.
__global__ void layer1_kernel(const float* __restrict__ robot,
                              const float* __restrict__ object,
                              const float* __restrict__ z,
                              const float* __restrict__ W1,
                              const float* __restrict__ b1,
                              uint32_t* __restrict__ hA16,
                              uint32_t* __restrict__ hB16) {
  int t = blockIdx.x * 256 + threadIdx.x;   // one thread = one f16 pair
  if (t >= B_ * N_ * KP_) return;
  int p  = t % KP_;
  int bn = t / KP_;
  int b  = bn / N_;
  int h0 = 2 * p;
  const float* ra = robot  + (size_t)bn * E_;
  const float* ob = object + (size_t)bn * E_;
  const float* zz = z + b * L_;
  float a0 = 0.f, a1 = 0.f, c0 = 0.f, c1 = 0.f;
  for (int d = 0; d < E_; ++d) {
    float va = ra[d], vb = ob[d];
    a0 += va * W1[(size_t)d * H_ + h0];
    a1 += va * W1[(size_t)d * H_ + h0 + 1];
    c0 += vb * W1[(size_t)(D_ + d) * H_ + h0];
    c1 += vb * W1[(size_t)(D_ + d) * H_ + h0 + 1];
  }
  for (int d = 0; d < L_; ++d) {
    float zv = zz[d];
    int dd = E_ + d;
    a0 += zv * W1[(size_t)dd * H_ + h0];
    a1 += zv * W1[(size_t)dd * H_ + h0 + 1];
    c0 += zv * W1[(size_t)(D_ + dd) * H_ + h0];
    c1 += zv * W1[(size_t)(D_ + dd) * H_ + h0 + 1];
  }
  c0 += b1[h0];
  c1 += b1[h0 + 1];
  PF pa; pa.h[0] = (_Float16)a0; pa.h[1] = (_Float16)a1;
  PF pb; pb.h[0] = (_Float16)c0; pb.h[1] = (_Float16)c1;
  hA16[t] = pa.u;
  hB16[t] = pb.u;
}

// relu(a+b) of 4+4 packed-f16 dwords into one A/B-fragment half
__device__ __forceinline__ void build_frag(v16h& av, const uint4& x0,
                                           const uint4& x1, const uint4& y0,
                                           const uint4& y1) {
  uint32_t ua[8] = {x0.x, x0.y, x0.z, x0.w, x1.x, x1.y, x1.z, x1.w};
  uint32_t ub[8] = {y0.x, y0.y, y0.z, y0.w, y1.x, y1.y, y1.z, y1.w};
#pragma unroll
  for (int v = 0; v < 8; ++v) {
    PF pa; pa.u = ua[v];
    PF pb; pb.u = ub[v];
    _Float16 s0 = pa.h[0] + pb.h[0];
    _Float16 s1 = pa.h[1] + pb.h[1];
    av[2 * v]     = (s0 > (_Float16)0) ? s0 : (_Float16)0;   // relu
    av[2 * v + 1] = (s1 > (_Float16)0) ? s1 : (_Float16)0;
  }
}

// --- kernel 2: pairwise layers 2+3 on WMMA.
// Wave-tile = (b, i0..i0+1, 16 j's).  A0/A1 (f16 16x32 frags, rows = j's)
// resident in 160 VGPRs; each B fragment (LDS) feeds two WMMAs.
__global__ __launch_bounds__(BLK)
void pairwise_mlp_kernel(const uint32_t* __restrict__ hA16,
                         const uint32_t* __restrict__ hB16,
                         const uint32_t* __restrict__ w2pk,
                         const float* __restrict__ b2,
                         const float* __restrict__ W3,
                         const float* __restrict__ b3,
                         float* __restrict__ out) {
  extern __shared__ uint32_t lds[];   // 200 KB: whole packed W2

  // cooperative global -> LDS stage (b128 copies)
  {
    const uint4* src = (const uint4*)w2pk;
    uint4*       dst = (uint4*)lds;
    for (int i = threadIdx.x; i < W2PACK_U32 / 4; i += BLK) dst[i] = src[i];
  }
  __syncthreads();

  const int lane = threadIdx.x & 31;
  const int wave = blockIdx.x * (BLK / 32) + (threadIdx.x >> 5);
  const int m    = lane & 15;   // row within 16-j tile / output column group
  const int hi   = lane >> 4;   // lane-half selects A K-subrange
  const float b3v = b3[0];

  for (int tt = 0; tt < TPW; ++tt) {
    int tile = wave * TPW + tt;          // exact cover: GRID*8*TPW == NTILES2
    int b    = tile >> 13;               // /(256*32)
    int rem  = tile & 8191;
    int i0   = (rem >> 5) * 2;           // i-pair base
    int jb   = rem & 31;
    int j    = jb * 16 + m;

    const uint4* Arow0 = (const uint4*)(hA16 + (size_t)(b * N_ + i0) * KP_);
    const uint4* Arow1 = Arow0 + (KP_ / 4);   // row i0+1
    const uint4* Brow  = (const uint4*)(hB16 + (size_t)(b * N_ + j) * KP_);

    // Build resident A fragments (rows = the 16 j's) for both i-rows.
    // K-pair pattern per ISA: v<4 -> K=(hi?8:0)+2v ; v>=4 -> K=(hi?24:16)+2(v-4).
    v16h A0[KT_], A1[KT_];
#pragma unroll
    for (int kt = 0; kt < KT_; ++kt) {
      uint4 e0 = Brow[kt * 4 + hi];       // hB[j] shared by both i-rows
      uint4 e1 = Brow[kt * 4 + 2 + hi];
      uint4 a0 = Arow0[kt * 4 + hi];
      uint4 a1 = Arow0[kt * 4 + 2 + hi];
      build_frag(A0[kt], a0, a1, e0, e1);
      uint4 d0 = Arow1[kt * 4 + hi];
      uint4 d1 = Arow1[kt * 4 + 2 + hi];
      build_frag(A1[kt], d0, d1, e0, e1);
    }

    float acc0[8], acc1[8];
#pragma unroll
    for (int r = 0; r < 8; ++r) { acc0[r] = 0.f; acc1[r] = 0.f; }

    for (int nt = 0; nt < NT_; ++nt) {
      float b2v = b2[nt * 16 + m];
      v8f c0 = {b2v, b2v, b2v, b2v, b2v, b2v, b2v, b2v};
      v8f c1 = c0;
      const uint32_t* bp = &lds[((nt * KT_) << 8) + lane * 8];
#pragma unroll
      for (int kt = 0; kt < KT_; ++kt) {
        struct U8 { uint4 a, b; } u;
        u.a = *(const uint4*)(bp + (kt << 8));      // ds_load_b128, imm offset
        u.b = *(const uint4*)(bp + (kt << 8) + 4);  // ds_load_b128, imm offset
        v16h Bv = __builtin_bit_cast(v16h, u);
        c0 = __builtin_amdgcn_wmma_f32_16x16x32_f16(
            false, A0[kt], false, Bv, (short)0, c0, false, false);
        c1 = __builtin_amdgcn_wmma_f32_16x16x32_f16(
            false, A1[kt], false, Bv, (short)0, c1, false, false);
      }
      // layer-3 fold: relu then scale by W3[col], col = nt*16 + m
      float w3v = W3[nt * 16 + m];
#pragma unroll
      for (int r = 0; r < 8; ++r) {
        acc0[r] += __builtin_fmaxf(c0[r], 0.f) * w3v;
        acc1[r] += __builtin_fmaxf(c1[r], 0.f) * w3v;
      }
    }

    // reduce over the 16 columns (lanes 0-15 and 16-31 independently)
#pragma unroll
    for (int r = 0; r < 8; ++r) {
      acc0[r] += __shfl_xor(acc0[r], 1, 32);
      acc0[r] += __shfl_xor(acc0[r], 2, 32);
      acc0[r] += __shfl_xor(acc0[r], 4, 32);
      acc0[r] += __shfl_xor(acc0[r], 8, 32);
      acc1[r] += __shfl_xor(acc1[r], 1, 32);
      acc1[r] += __shfl_xor(acc1[r], 2, 32);
      acc1[r] += __shfl_xor(acc1[r], 4, 32);
      acc1[r] += __shfl_xor(acc1[r], 8, 32);
    }

    float* orow0 = out + (size_t)(b * N_ + i0) * N_ + jb * 16;
    float* orow1 = orow0 + N_;
    if (lane == 0) {            // rows (j offsets) 0..7 live in VGPR r, lanes 0-15
#pragma unroll
      for (int r = 0; r < 8; ++r) { orow0[r] = acc0[r] + b3v; orow1[r] = acc1[r] + b3v; }
    } else if (lane == 16) {    // rows 8..15 live in VGPR r, lanes 16-31
#pragma unroll
      for (int r = 0; r < 8; ++r) { orow0[8 + r] = acc0[r] + b3v; orow1[8 + r] = acc1[r] + b3v; }
    }
  }
}

// ---------------------------------------------------------------------------
// Inputs (setup_inputs order): robot[2,512,128] object[2,512,128] z[2,32]
// W1[320,320] b1[320] W2[320,320] b2[320] W3[320,1] b3[1].  Output f32 [2,512,512].
// Workspace: hA16 (640KB) | hB16 (640KB) | w2pk (200KB)  ~1.5 MB total.
// ---------------------------------------------------------------------------
extern "C" void kernel_launch(void* const* d_in, const int* in_sizes, int n_in,
                              void* d_out, int out_size, void* d_ws, size_t ws_size,
                              hipStream_t stream) {
  const float* robot  = (const float*)d_in[0];
  const float* object = (const float*)d_in[1];
  const float* z      = (const float*)d_in[2];
  const float* W1     = (const float*)d_in[3];
  const float* b1     = (const float*)d_in[4];
  const float* W2     = (const float*)d_in[5];
  const float* b2     = (const float*)d_in[6];
  const float* W3     = (const float*)d_in[7];
  const float* b3     = (const float*)d_in[8];
  float* out = (float*)d_out;

  uint32_t* ws   = (uint32_t*)d_ws;
  uint32_t* hA16 = ws;
  uint32_t* hB16 = ws + HAB;
  uint32_t* w2pk = ws + 2 * HAB;

  pack_w2_kernel<<<(W2PACK_U32 + 255) / 256, 256, 0, stream>>>(W2, w2pk);
  layer1_kernel<<<(B_ * N_ * KP_ + 255) / 256, 256, 0, stream>>>(
      robot, object, z, W1, b1, hA16, hB16);
  pairwise_mlp_kernel<<<GRID, BLK, W2PACK_U32 * 4, stream>>>(
      hA16, hB16, w2pk, b2, W3, b3, out);
}
